// ArithmeticAttention_30305289241249
// MI455X (gfx1250) — matrix operand order repaired and checked by reading, not compiled, not run on hardware
//
#include <hip/hip_runtime.h>
#include <hip/hip_bf16.h>

// ---------------------------------------------------------------------------
// ArithmeticAttention for MI455X (gfx1250, wave32, WMMA bf16 path)
// B=2, S=2048, D=1024, H=16, HD=64
// GEMMs: 32x64 tile per wave, 8 independent v_wmma accumulator chains.
// Attention: flash-style, bias matrix precomputed once (L2-resident, 16x reuse).
// ---------------------------------------------------------------------------

typedef __attribute__((ext_vector_type(16))) __bf16 bf16x16;
typedef __attribute__((ext_vector_type(8)))  __bf16 bf16x8;
typedef __attribute__((ext_vector_type(8)))  float  floatx8;

constexpr int Bz  = 2;
constexpr int Sz  = 2048;
constexpr int Dz  = 1024;
constexpr int Hz  = 16;
constexpr int HDz = 64;
constexpr int Mz  = Bz * Sz;              // 4096 rows for all GEMMs
constexpr float SCALE = 0.125f;           // 1/sqrt(64)

// ---- helpers --------------------------------------------------------------

__device__ inline __bf16 to_bf16(float f) {
  unsigned u = __builtin_bit_cast(unsigned, f);
  u += 0x7FFFu + ((u >> 16) & 1u);        // round-to-nearest-even
  unsigned short h = (unsigned short)(u >> 16);
  return __builtin_bit_cast(__bf16, h);
}

__device__ inline floatx8 wmma_bf16(bf16x16 a, bf16x16 b, floatx8 c) {
  return __builtin_amdgcn_wmma_f32_16x16x32_bf16(
      /*neg_a=*/false, a, /*neg_b=*/false, b,
      /*c_mod=*/(short)0, c, /*reuse_a=*/false, /*reuse_b=*/false);
}

// A-matrix 16x32 bf16 operand: lane<16 holds K={0..7,16..23}, lane>=16 holds
// K={8..15,24..31} of row M=lane&15 (ISA 7.12.2). rowk -> row base + k-chunk.
__device__ inline bf16x16 load_a16(const __bf16* rowk, int half) {
  bf16x8 u0 = *(const bf16x8*)(rowk + half * 8);
  bf16x8 u1 = *(const bf16x8*)(rowk + 16 + half * 8);
  bf16x16 r;
#pragma unroll
  for (int i = 0; i < 8; ++i) { r[i] = u0[i]; r[i + 8] = u1[i]; }
  return r;
}

// B-matrix 32x16 bf16 operand: lanes 0-15 hold K={0..7,16..23}, lanes 16-31
// hold K={8..15,24..31}, of column N=lane&15 (matches A). rowk points at
// the K-major row for this lane's N.
__device__ inline bf16x16 load_b16(const __bf16* rowk, int half) {
  bf16x8 u0 = *(const bf16x8*)(rowk + half * 8);
  bf16x8 u1 = *(const bf16x8*)(rowk + 16 + half * 8);
  bf16x16 r;
#pragma unroll
  for (int i = 0; i < 8; ++i) { r[i] = u0[i]; r[i + 8] = u1[i]; }
  return r;
}

// ---- kernel 1: f32 -> bf16 convert ----------------------------------------

__global__ __launch_bounds__(256) void cvt_bf16_kernel(
    const float* __restrict__ src, __bf16* __restrict__ dst, int n) {
  int i = blockIdx.x * blockDim.x + threadIdx.x;
  if (i < n) dst[i] = to_bf16(src[i]);
}

// ---- kernel 2: arithmetic bias precompute (B,S,S) f32 ---------------------
// Head-independent: computed once, reused by all 16 heads out of L2
// (33.5 MB << 192 MB L2). Saves 15/16 of the u32-division VALU work.

__global__ __launch_bounds__(256) void abias_kernel(
    const int* __restrict__ enc, float* __restrict__ out) {
  int idx = blockIdx.x * blockDim.x + threadIdx.x;   // < Bz*Sz*Sz by grid
  int b  = idx / (Sz * Sz);
  int r  = idx - b * (Sz * Sz);
  int qi = r / Sz;
  int kj = r - qi * Sz;
  unsigned ei = (unsigned)enc[b * Sz + qi];
  unsigned ej = (unsigned)enc[b * Sz + kj];
  float v = 0.0f;
  if (qi != kj) {
    unsigned m1 = ei % (ej + 1u);
    unsigned m2 = ej % (ei + 1u);
    unsigned g  = (m1 < m2) ? m1 : m2;
    unsigned mx = (ei > ej) ? ei : ej;
    v = (float)g / (float)(mx + 1u);
  }
  out[idx] = v;
}

// ---- kernel 3: WMMA GEMM  C[M,N] = A[M,K] @ W[N,K]^T + bias ---------------
// 32(M) x 64(N) per wave: 8 independent accumulator chains; A operands
// reused across 4 N-subtiles, B operands across 2 M-subtiles.
// outF != nullptr  -> store f32 to outF[m*N+n]          (final projection)
// else tS == 0     -> store bf16 to outB[m*N+n]         (Q, K)
// else tS == Sz    -> store bf16 transposed (b,d,s)     (V^T for attention)

__device__ inline void store_tile16(floatx8 cc, float bv, int mbase, int n,
                                    int half, __bf16* __restrict__ outB,
                                    float* __restrict__ outF, int N, int tS) {
#pragma unroll
  for (int v = 0; v < 8; ++v) {
    float val = cc[v] + bv;
    int m = mbase + v + half * 8;
    if (outF) {
      outF[(size_t)m * N + n] = val;
    } else if (tS > 0) {
      int bi = m / tS, srow = m - bi * tS;       // (b, d, s) layout
      outB[((size_t)bi * N + n) * tS + srow] = to_bf16(val);
    } else {
      outB[(size_t)m * N + n] = to_bf16(val);
    }
  }
}

__global__ __launch_bounds__(256) void gemm_wmma_kernel(
    const __bf16* __restrict__ A, const __bf16* __restrict__ W,
    const float* __restrict__ bias, __bf16* __restrict__ outB,
    float* __restrict__ outF, int M, int N, int K, int tS) {
  int wid  = (blockIdx.x * blockDim.x + threadIdx.x) >> 5;
  int lane = threadIdx.x & 31;
  int half = lane >> 4;
  int lm   = lane & 15;
  int tilesN = N >> 6;                       // 64-wide N tiles
  int tm = wid / tilesN;
  int tn = wid - tm * tilesN;
  int m0 = tm * 32;
  int n0 = tn * 64;
  if (m0 >= M) return;

  const __bf16* ar0 = A + (size_t)(m0 + lm) * K;
  const __bf16* ar1 = A + (size_t)(m0 + 16 + lm) * K;
  const __bf16* wr0 = W + (size_t)(n0 + 0 * 16 + lm) * K;
  const __bf16* wr1 = W + (size_t)(n0 + 1 * 16 + lm) * K;
  const __bf16* wr2 = W + (size_t)(n0 + 2 * 16 + lm) * K;
  const __bf16* wr3 = W + (size_t)(n0 + 3 * 16 + lm) * K;

  floatx8 c00 = {}, c01 = {}, c02 = {}, c03 = {};
  floatx8 c10 = {}, c11 = {}, c12 = {}, c13 = {};

  for (int kc = 0; kc < K; kc += 32) {
    // keep the streams warm a few chunks ahead (global_prefetch_b8)
    __builtin_prefetch(ar0 + kc + 128, 0, 1);
    __builtin_prefetch(ar1 + kc + 128, 0, 1);
    __builtin_prefetch(wr0 + kc + 128, 0, 1);
    __builtin_prefetch(wr2 + kc + 128, 0, 1);

    bf16x16 a0 = load_a16(ar0 + kc, half);
    bf16x16 a1 = load_a16(ar1 + kc, half);
    bf16x16 b0 = load_b16(wr0 + kc, half);
    c00 = wmma_bf16(a0, b0, c00);
    c10 = wmma_bf16(a1, b0, c10);
    bf16x16 b1 = load_b16(wr1 + kc, half);
    c01 = wmma_bf16(a0, b1, c01);
    c11 = wmma_bf16(a1, b1, c11);
    bf16x16 b2 = load_b16(wr2 + kc, half);
    c02 = wmma_bf16(a0, b2, c02);
    c12 = wmma_bf16(a1, b2, c12);
    bf16x16 b3 = load_b16(wr3 + kc, half);
    c03 = wmma_bf16(a0, b3, c03);
    c13 = wmma_bf16(a1, b3, c13);
  }

  float bv0 = bias[n0 + 0 * 16 + lm];
  float bv1 = bias[n0 + 1 * 16 + lm];
  float bv2 = bias[n0 + 2 * 16 + lm];
  float bv3 = bias[n0 + 3 * 16 + lm];
  store_tile16(c00, bv0, m0, n0 + 0 * 16 + lm, half, outB, outF, N, tS);
  store_tile16(c01, bv1, m0, n0 + 1 * 16 + lm, half, outB, outF, N, tS);
  store_tile16(c02, bv2, m0, n0 + 2 * 16 + lm, half, outB, outF, N, tS);
  store_tile16(c03, bv3, m0, n0 + 3 * 16 + lm, half, outB, outF, N, tS);
  store_tile16(c10, bv0, m0 + 16, n0 + 0 * 16 + lm, half, outB, outF, N, tS);
  store_tile16(c11, bv1, m0 + 16, n0 + 1 * 16 + lm, half, outB, outF, N, tS);
  store_tile16(c12, bv2, m0 + 16, n0 + 2 * 16 + lm, half, outB, outF, N, tS);
  store_tile16(c13, bv3, m0 + 16, n0 + 3 * 16 + lm, half, outB, outF, N, tS);
}

// ---- kernel 4: flash attention --------------------------------------------
// One wave per (b, h, 16-row q-tile). KV processed in chunks of 32.
// Q, K in (b,s,d) bf16; V transposed (b,d,s) bf16; bias matrix f32 in L2.

__global__ __launch_bounds__(256) void attn_wmma_kernel(
    const __bf16* __restrict__ Q, const __bf16* __restrict__ Kmat,
    const __bf16* __restrict__ Vt, const float* __restrict__ biasmat,
    const float* __restrict__ head_bias, const float* __restrict__ gcdw,
    __bf16* __restrict__ out) {
  __shared__ __bf16 lsh[8][16 * 32];                 // per-wave P scratch

  int wib  = threadIdx.x >> 5;
  int wid  = (blockIdx.x * blockDim.x + threadIdx.x) >> 5;
  int lane = threadIdx.x & 31;
  int half = lane >> 4;
  int lm   = lane & 15;

  int qt = wid & 127;            // S/16 = 128 q-tiles
  int bh = wid >> 7;
  int h  = bh & 15;
  int b  = bh >> 4;
  int q0 = qt * 16;

  // Q tile as two A operands (d = 0..31, 32..63)
  const __bf16* qbase = Q + ((size_t)(b * Sz + q0 + lm) * Dz + h * HDz);
  bf16x16 aq0 = load_a16(qbase, half);
  bf16x16 aq1 = load_a16(qbase + 32, half);

  const float gw = gcdw[0];
  const float hb = head_bias[h];
  const float* brow = biasmat + (size_t)b * Sz * Sz;

  float  mrow[8], lrow[8];
  floatx8 o0 = {}, o1 = {}, o2 = {}, o3 = {};
#pragma unroll
  for (int v = 0; v < 8; ++v) { mrow[v] = -1e30f; lrow[v] = 0.0f; }

  __bf16* l = &lsh[wib][0];

  for (int j0 = 0; j0 < Sz; j0 += 32) {
    // ---- scores: two 16x16 tiles (kv columns j0..j0+15, j0+16..j0+31)
    const __bf16* kr0 = Kmat + ((size_t)(b * Sz + j0 + lm) * Dz + h * HDz);
    const __bf16* kr1 = Kmat + ((size_t)(b * Sz + j0 + 16 + lm) * Dz + h * HDz);
    floatx8 s0 = {}, s1 = {};
    s0 = wmma_bf16(aq0, load_b16(kr0, half), s0);
    s0 = wmma_bf16(aq1, load_b16(kr0 + 32, half), s0);
    s1 = wmma_bf16(aq0, load_b16(kr1, half), s1);
    s1 = wmma_bf16(aq1, load_b16(kr1 + 32, half), s1);

    // ---- bias + online softmax (rows live per half-wave; 16-lane reduce)
#pragma unroll
    for (int v = 0; v < 8; ++v) {
      int qi = q0 + v + half * 8;
      const float* bq = brow + (size_t)qi * Sz + j0;
      float x0 = s0[v] * SCALE + bq[lm] * gw + hb;
      float x1 = s1[v] * SCALE + bq[16 + lm] * gw + hb;
      float mx = fmaxf(fmaxf(x0, x1), mrow[v]);
#pragma unroll
      for (int off = 1; off < 16; off <<= 1)
        mx = fmaxf(mx, __shfl_xor(mx, off, 32));
      float alpha = __expf(mrow[v] - mx);
      mrow[v] = mx;
      float p0 = __expf(x0 - mx);
      float p1 = __expf(x1 - mx);
      float rs = p0 + p1;
#pragma unroll
      for (int off = 1; off < 16; off <<= 1)
        rs += __shfl_xor(rs, off, 32);
      lrow[v] = lrow[v] * alpha + rs;
      o0[v] *= alpha; o1[v] *= alpha; o2[v] *= alpha; o3[v] *= alpha;
      // write P (C-layout element (m,n)) to LDS row-major 16x32 bf16
      int m = v + half * 8;
      l[m * 32 + lm]      = to_bf16(p0);
      l[m * 32 + 16 + lm] = to_bf16(p1);
    }

    // per-wave LDS: DS ops are in-order within a wave; wait for stores
    asm volatile("s_wait_dscnt 0" ::: "memory");

    // ---- re-read P in A-operand layout (16x32, K = kv chunk)
    const __bf16* lr = l + lm * 32;
    bf16x16 ap = load_a16(lr, half);

    // ---- O += P @ V_chunk ; V^T rows are contiguous along kv (K dim)
    const __bf16* vtb =
        Vt + ((size_t)b * Dz + h * HDz) * Sz + j0;   // + d*Sz per column
    o0 = wmma_bf16(ap, load_b16(vtb + (size_t)(0 * 16 + lm) * Sz, half), o0);
    o1 = wmma_bf16(ap, load_b16(vtb + (size_t)(1 * 16 + lm) * Sz, half), o1);
    o2 = wmma_bf16(ap, load_b16(vtb + (size_t)(2 * 16 + lm) * Sz, half), o2);
    o3 = wmma_bf16(ap, load_b16(vtb + (size_t)(3 * 16 + lm) * Sz, half), o3);
  }

  // ---- normalize and store attended (b, s, h*64+d) as bf16
#pragma unroll
  for (int v = 0; v < 8; ++v) {
    float inv = 1.0f / lrow[v];
    int srow = q0 + v + half * 8;
    size_t base = ((size_t)(b * Sz + srow)) * Dz + h * HDz + lm;
    out[base + 0]  = to_bf16(o0[v] * inv);
    out[base + 16] = to_bf16(o1[v] * inv);
    out[base + 32] = to_bf16(o2[v] * inv);
    out[base + 48] = to_bf16(o3[v] * inv);
  }
}

// ---- launch ---------------------------------------------------------------

extern "C" void kernel_launch(void* const* d_in, const int* in_sizes, int n_in,
                              void* d_out, int out_size, void* d_ws, size_t ws_size,
                              hipStream_t stream) {
  const float* hidden = (const float*)d_in[0];
  const int*   enc    = (const int*)d_in[1];
  const float* Wq = (const float*)d_in[2];
  const float* bq = (const float*)d_in[3];
  const float* Wk = (const float*)d_in[4];
  const float* bk = (const float*)d_in[5];
  const float* Wv = (const float*)d_in[6];
  const float* bv = (const float*)d_in[7];
  const float* Wo = (const float*)d_in[8];
  const float* bo = (const float*)d_in[9];
  const float* ahb  = (const float*)d_in[10];
  const float* gcdw = (const float*)d_in[11];
  float* out = (float*)d_out;

  // workspace layout (bytes); total ~84 MB
  char* ws = (char*)d_ws;
  __bf16* HID = (__bf16*)(ws + 0);                    //  8 MB hidden bf16
  __bf16* WQb = (__bf16*)(ws + (8u << 20));           //  2 MB
  __bf16* WKb = (__bf16*)(ws + (10u << 20));          //  2 MB
  __bf16* WVb = (__bf16*)(ws + (12u << 20));          //  2 MB
  __bf16* WOb = (__bf16*)(ws + (14u << 20));          //  2 MB
  __bf16* QB  = (__bf16*)(ws + (16u << 20));          //  8 MB
  __bf16* KB  = (__bf16*)(ws + (24u << 20));          //  8 MB
  __bf16* VT  = (__bf16*)(ws + (32u << 20));          //  8 MB (b,d,s)
  __bf16* ATT = (__bf16*)(ws + (40u << 20));          //  8 MB attended
  float*  BIAS = (float*)(ws + (48u << 20));          // 32 MB (B,S,S)

  const int nHid = Bz * Sz * Dz;                      // 4194304
  const int nW   = Dz * Dz;                           // 1048576

  cvt_bf16_kernel<<<(nHid + 255) / 256, 256, 0, stream>>>(hidden, HID, nHid);
  cvt_bf16_kernel<<<(nW + 255) / 256, 256, 0, stream>>>(Wq, WQb, nW);
  cvt_bf16_kernel<<<(nW + 255) / 256, 256, 0, stream>>>(Wk, WKb, nW);
  cvt_bf16_kernel<<<(nW + 255) / 256, 256, 0, stream>>>(Wv, WVb, nW);
  cvt_bf16_kernel<<<(nW + 255) / 256, 256, 0, stream>>>(Wo, WOb, nW);

  abias_kernel<<<(Bz * Sz * Sz) / 256, 256, 0, stream>>>(enc, BIAS);

  // 32x64 tiles: (M/32)*(N/64) waves, 8 waves per 256-thread block
  const int gemmBlocks = (Mz / 32) * (Dz / 64) / 8;   // 256
  gemm_wmma_kernel<<<gemmBlocks, 256, 0, stream>>>(HID, WQb, bq, QB, nullptr,
                                                   Mz, Dz, Dz, 0);
  gemm_wmma_kernel<<<gemmBlocks, 256, 0, stream>>>(HID, WKb, bk, KB, nullptr,
                                                   Mz, Dz, Dz, 0);
  gemm_wmma_kernel<<<gemmBlocks, 256, 0, stream>>>(HID, WVb, bv, VT, nullptr,
                                                   Mz, Dz, Dz, Sz);

  const int attnBlocks = Bz * Hz * (Sz / 16) / 8;     // 4096 waves / 8
  attn_wmma_kernel<<<attnBlocks, 256, 0, stream>>>(QB, KB, VT, BIAS, ahb, gcdw,
                                                   ATT);

  gemm_wmma_kernel<<<gemmBlocks, 256, 0, stream>>>(ATT, WOb, bo, nullptr, out,
                                                   Mz, Dz, Dz, 0);
}
